// CausalSpaceSelfAttention_27943057227875
// MI455X (gfx1250) — compile-verified, matrix operand order checked by reading, not run on hardware
//
#include <hip/hip_runtime.h>

// ---------------------------------------------------------------------------
// CDNA5 (gfx1250) wave32 WMMA implementation of causal self-attention with
// QK-layernorm and axial rotary embeddings.  All GEMMs use
// v_wmma_f32_16x16x32_bf16; attention is flash-style with online softmax.
// Round 2: software-pipelined (double-buffered) GEMM K-loop so fragment
// loads for step k+32 overlap the WMMAs of step k instead of draining
// LOADcnt to 0 before every matrix op.
// ---------------------------------------------------------------------------

typedef __attribute__((ext_vector_type(16))) __bf16 bfx16;
typedef __attribute__((ext_vector_type(8)))  __bf16 bfx8;
typedef __attribute__((ext_vector_type(8)))  float  fx8;

union BF16Frag { bfx16 v; bfx8 h[2]; };

// Per the CDNA5 ISA 16-bit fragment layout (A and B identical; B's "row" is
// the output column):
//   lane<16 : elems j=0..7 -> K = k0+j,      j=8..15 -> K = k0+16+(j-8)
//   lane>=16: same +8
// => two contiguous 16-byte chunks at k0+8*hf and k0+8*hf+16.

// Generic: row chosen per call.
__device__ __forceinline__ bfx16 load_frag16x32(const __bf16* __restrict__ base,
                                                int stride, int row, int k0, int hf) {
  const __bf16* p = base + (size_t)row * stride + k0 + hf * 8;
  BF16Frag f;
  f.h[0] = *(const bfx8*)(p);
  f.h[1] = *(const bfx8*)(p + 16);
  return f.v;
}

// Fast path: lane's row base (incl. hf*8) precomputed; only k0 varies.
__device__ __forceinline__ bfx16 load_frag_at(const __bf16* __restrict__ p, int k0) {
  BF16Frag f;
  f.h[0] = *(const bfx8*)(p + k0);
  f.h[1] = *(const bfx8*)(p + k0 + 16);
  return f.v;
}

__device__ __forceinline__ fx8 wmma_bf16(bfx16 a, bfx16 b, fx8 c) {
  return __builtin_amdgcn_wmma_f32_16x16x32_bf16(false, a, false, b, (short)0, c,
                                                 false, false);
}

// ---------------------------------------------------------------------------
// fp32 -> bf16 elementwise convert
// ---------------------------------------------------------------------------
__global__ void cvt_f32_to_bf16(const float* __restrict__ in,
                                __bf16* __restrict__ out, size_t n) {
  size_t i = (size_t)blockIdx.x * blockDim.x + threadIdx.x;
  size_t stride = (size_t)gridDim.x * blockDim.x;
  for (; i < n; i += stride) out[i] = (__bf16)in[i];
}

// ---------------------------------------------------------------------------
// C[M,N] (f32) = A[M,K] (bf16, row-major) @ B[N,K]^T (bf16, row-major = W)
// One wave computes a 16x64 tile.  Double-buffered K loop: fragments for
// k0+32 are issued before the WMMAs consuming the k0 fragments.
// M multiple of 16, N of 64, K of 32 (holds for all uses here).
// ---------------------------------------------------------------------------
__global__ void gemm_bf16_wmma(const __bf16* __restrict__ A,
                               const __bf16* __restrict__ B,
                               float* __restrict__ C,
                               int M, int N, int K) {
  int wave = (int)((blockIdx.x * blockDim.x + threadIdx.x) >> 5);
  int lane = threadIdx.x & 31;
  int hf   = lane >> 4;
  int l16  = lane & 15;

  int nQuads = N >> 6;
  int mTile  = wave / nQuads;
  int nQuad  = wave - mTile * nQuads;
  if (mTile >= (M >> 4)) return;

  int n0 = nQuad * 64;

  // per-lane row base pointers (include hf*8 element K-offset)
  const __bf16* aP = A + (size_t)(mTile * 16 + l16) * K + hf * 8;
  const __bf16* bP[4];
#pragma unroll
  for (int g = 0; g < 4; ++g)
    bP[g] = B + (size_t)(n0 + g * 16 + l16) * K + hf * 8;

  fx8 acc[4] = {};

  // prologue: load first K-step fragments
  bfx16 aN = load_frag_at(aP, 0);
  bfx16 bN[4];
#pragma unroll
  for (int g = 0; g < 4; ++g) bN[g] = load_frag_at(bP[g], 0);

  for (int k0 = 0; k0 < K; k0 += 32) {
    bfx16 aC = aN;
    bfx16 bC[4];
#pragma unroll
    for (int g = 0; g < 4; ++g) bC[g] = bN[g];

    int kn = k0 + 32;
    if (kn < K) {
      __builtin_prefetch(aP + kn + 256, 0, 1);
      aN = load_frag_at(aP, kn);
#pragma unroll
      for (int g = 0; g < 4; ++g) {
        __builtin_prefetch(bP[g] + kn + 256, 0, 1);
        bN[g] = load_frag_at(bP[g], kn);
      }
    }

#pragma unroll
    for (int g = 0; g < 4; ++g) acc[g] = wmma_bf16(aC, bC[g], acc[g]);
  }

#pragma unroll
  for (int g = 0; g < 4; ++g) {
#pragma unroll
    for (int v = 0; v < 8; ++v) {
      int m = mTile * 16 + v + 8 * hf;
      C[(size_t)m * N + n0 + g * 16 + l16] = acc[g][v];
    }
  }
}

// ---------------------------------------------------------------------------
// Fused LayerNorm + axial rotary.  One 256-thread block per row (C=1024,
// 4 elems/thread => 2 rotary pairs/thread).  Output is bf16 in (B,H,T,D).
// ---------------------------------------------------------------------------
__global__ void ln_rope_kernel(const float* __restrict__ Z,   // (B*T, C)
                               const float* __restrict__ gam,
                               const float* __restrict__ bet,
                               __bf16* __restrict__ Out,      // (B,H,T,D)
                               int B, int T, int C) {
  const int H = 16, D = 64, PREFIX = 6, ENDX = 18;
  int rowIdx = blockIdx.x;
  int b = rowIdx / T, t = rowIdx - b * T;
  const float* z = Z + (size_t)rowIdx * C;

  int tid = threadIdx.x;
  int c0 = tid * 4;
  float4 vz = *(const float4*)(z + c0);

  float s  = vz.x + vz.y + vz.z + vz.w;
  float sq = vz.x * vz.x + vz.y * vz.y + vz.z * vz.z + vz.w * vz.w;
#pragma unroll
  for (int off = 16; off >= 1; off >>= 1) {
    s  += __shfl_xor(s, off, 32);
    sq += __shfl_xor(sq, off, 32);
  }
  __shared__ float red[16];
  int w = tid >> 5;
  if ((tid & 31) == 0) { red[w] = s; red[8 + w] = sq; }
  __syncthreads();
  float S = 0.f, SQ = 0.f;
#pragma unroll
  for (int i = 0; i < 8; ++i) { S += red[i]; SQ += red[8 + i]; }
  float mean = S / (float)C;
  float var  = SQ / (float)C - mean * mean;
  float inv  = rsqrtf(var + 1e-5f);

  float zz[4] = {vz.x, vz.y, vz.z, vz.w};
  float o[4];
#pragma unroll
  for (int i = 0; i < 4; ++i) {
    int c = c0 + i;
    o[i] = (zz[i] - mean) * inv * gam[c] + bet[c];
  }

  if (t >= PREFIX) {
    int L = t - PREFIX;
    float tx = (float)(L % ENDX);
    float ty = (float)(L / ENDX);
    int d0 = c0 & (D - 1);
#pragma unroll
    for (int pp = 0; pp < 2; ++pp) {
      int p = (d0 >> 1) + pp;        // rotary pair index 0..31
      int j = p & 15;                // frequency index
      float coeff = (p < 16) ? tx : ty;
      float fr = powf(1000.0f, -(float)j * (1.0f / 16.0f));
      float ang = coeff * fr;
      float cs = cosf(ang), sn = sinf(ang);
      float xe = o[2 * pp], xo = o[2 * pp + 1];
      o[2 * pp]     = xe * cs - xo * sn;
      o[2 * pp + 1] = xe * sn + xo * cs;
    }
  }

  int h = c0 >> 6;
  size_t outBase = (((size_t)b * H + h) * T + t) * D + (c0 & (D - 1));
#pragma unroll
  for (int i = 0; i < 4; ++i) Out[outBase + i] = (__bf16)o[i];
}

// ---------------------------------------------------------------------------
// v_pre (B*T, C) f32 -> Vt (B,H,D,T) bf16  (so P@V B-fragments load contiguous)
// ---------------------------------------------------------------------------
__global__ void v_transpose_kernel(const float* __restrict__ Vp,
                                   __bf16* __restrict__ Vt,
                                   int B, int T, int C) {
  const int H = 16, D = 64;
  size_t n = (size_t)B * T * C;
  size_t i = (size_t)blockIdx.x * blockDim.x + threadIdx.x;
  size_t stride = (size_t)gridDim.x * blockDim.x;
  for (; i < n; i += stride) {
    int c = (int)(i % C);
    size_t r = i / C;
    int t = (int)(r % T);
    int b = (int)(r / T);
    int h = c >> 6, d = c & (D - 1);
    Vt[(((size_t)b * H + h) * D + d) * T + t] = (__bf16)Vp[i];
  }
}

// ---------------------------------------------------------------------------
// Flash attention: one wave (32 threads) per (b, h, 16-row q-tile).
// Scores via WMMA bf16 (K = head dim 64 -> 2 steps), online softmax with
// intra-half shuffle reductions, P transposed D-layout -> A-layout through
// LDS, P@V via WMMA bf16 (K = 32 sequence positions per step).
// ---------------------------------------------------------------------------
__global__ void attn_kernel(const __bf16* __restrict__ Qh,   // (B,H,T,D)
                            const __bf16* __restrict__ Kh,   // (B,H,T,D)
                            const __bf16* __restrict__ Vt,   // (B,H,D,T)
                            const float*  __restrict__ mask, // (B,T,T)
                            __bf16* __restrict__ Ybt,        // (B*T, C)
                            int B, int T) {
  const int H = 16, D = 64, C = 1024;
  int nQT = (T + 15) >> 4;
  int bid = blockIdx.x;
  int qt = bid % nQT;
  int bh = bid / nQT;
  int h = bh % H, b = bh / H;

  int lane = threadIdx.x & 31;
  int hf = lane >> 4, l16 = lane & 15;
  int q0 = qt * 16;
  int qrow = q0 + l16;
  int qc = qrow < T ? qrow : T - 1;

  const __bf16* Qb = Qh + ((size_t)b * H + h) * T * D;
  const __bf16* Kb = Kh + ((size_t)b * H + h) * T * D;
  const __bf16* Vb = Vt + ((size_t)b * H + h) * D * T;

  bfx16 aQ0 = load_frag16x32(Qb, D, qc, 0, hf);
  bfx16 aQ1 = load_frag16x32(Qb, D, qc, 32, hf);

  float rowM[8], rowL[8];
#pragma unroll
  for (int v = 0; v < 8; ++v) { rowM[v] = -1e30f; rowL[v] = 0.f; }
  fx8 acc[4] = {};
  const float scale = 0.125f;  // 1/sqrt(64)

  __shared__ __align__(16) __bf16 ptile[16 * 32];

  int kend = q0 + 16 < T ? q0 + 16 : T;  // causal upper bound (exclusive)
  for (int k0 = 0; k0 < kend; k0 += 32) {
    fx8 sv[2];
#pragma unroll
    for (int s = 0; s < 2; ++s) {
      int kb = k0 + s * 16;
      int krow = kb + l16;
      int kc = krow < T ? krow : T - 1;
      bfx16 bK0 = load_frag16x32(Kb, D, kc, 0, hf);
      bfx16 bK1 = load_frag16x32(Kb, D, kc, 32, hf);
      fx8 sc = {};
      sc = wmma_bf16(aQ0, bK0, sc);
      sc = wmma_bf16(aQ1, bK1, sc);
#pragma unroll
      for (int v = 0; v < 8; ++v) {
        int qq = q0 + v + 8 * hf;
        int kk = kb + l16;
        float mv = (qq < T && kk < T) ? mask[((size_t)b * T + qq) * T + kk]
                                      : -1e9f;
        sv[s][v] = sc[v] * scale + mv;
      }
    }
    // ---- online softmax (rows live in one 16-lane half each) ----
    float corr[8];
#pragma unroll
    for (int v = 0; v < 8; ++v) {
      float mx = fmaxf(sv[0][v], sv[1][v]);
#pragma unroll
      for (int off = 1; off < 16; off <<= 1) mx = fmaxf(mx, __shfl_xor(mx, off, 32));
      float newM = fmaxf(rowM[v], mx);
      float c = expf(rowM[v] - newM);
      float p0 = expf(sv[0][v] - newM);
      float p1 = expf(sv[1][v] - newM);
      sv[0][v] = p0; sv[1][v] = p1;
      float rs = p0 + p1;
#pragma unroll
      for (int off = 1; off < 16; off <<= 1) rs += __shfl_xor(rs, off, 32);
      rowL[v] = rowL[v] * c + rs;
      rowM[v] = newM;
      corr[v] = c;
    }
#pragma unroll
    for (int dt = 0; dt < 4; ++dt)
#pragma unroll
      for (int v = 0; v < 8; ++v) acc[dt][v] *= corr[v];

    // ---- transpose P (D-layout f32) -> A-layout bf16 via LDS ----
#pragma unroll
    for (int s = 0; s < 2; ++s)
#pragma unroll
      for (int v = 0; v < 8; ++v)
        ptile[(v + 8 * hf) * 32 + s * 16 + l16] = (__bf16)sv[s][v];

    BF16Frag pf;
    const __bf16* pp = &ptile[l16 * 32 + hf * 8];
    pf.h[0] = *(const bfx8*)(pp);
    pf.h[1] = *(const bfx8*)(pp + 16);

    // ---- P @ V ----
#pragma unroll
    for (int dt = 0; dt < 4; ++dt) {
      bfx16 bV = load_frag16x32(Vb, T, dt * 16 + l16, k0, hf);
      acc[dt] = wmma_bf16(pf.v, bV, acc[dt]);
    }
  }

  // ---- epilogue: normalize and store y in (B,T,C) bf16 ----
#pragma unroll
  for (int dt = 0; dt < 4; ++dt) {
#pragma unroll
    for (int v = 0; v < 8; ++v) {
      int qq = q0 + v + 8 * hf;
      if (qq < T)
        Ybt[((size_t)(b * T + qq)) * C + h * D + dt * 16 + l16] =
            (__bf16)(acc[dt][v] / rowL[v]);
    }
  }
}

// ---------------------------------------------------------------------------
// Host-side orchestration
// ---------------------------------------------------------------------------
extern "C" void kernel_launch(void* const* d_in, const int* in_sizes, int n_in,
                              void* d_out, int out_size, void* d_ws, size_t ws_size,
                              hipStream_t stream) {
  (void)in_sizes; (void)n_in; (void)out_size; (void)ws_size;
  const int B = 16, T = 582, C = 1024;
  const size_t M = (size_t)B * T;                 // 9312 (multiple of 16)

  const float* x    = (const float*)d_in[0];
  const float* amask= (const float*)d_in[1];
  const float* Wq   = (const float*)d_in[2];
  const float* Wk   = (const float*)d_in[3];
  const float* Wv   = (const float*)d_in[4];
  const float* Wo   = (const float*)d_in[5];
  const float* qg   = (const float*)d_in[6];
  const float* qb   = (const float*)d_in[7];
  const float* kg   = (const float*)d_in[8];
  const float* kb   = (const float*)d_in[9];
  float* out = (float*)d_out;

  // workspace layout
  char* ws = (char*)d_ws;
  size_t off = 0;
  auto alloc = [&](size_t bytes) {
    off = (off + 255) & ~(size_t)255;
    size_t o = off; off += bytes; return o;
  };
  const size_t szMC_bf = M * C * sizeof(__bf16);
  const size_t szMC_f  = M * C * sizeof(float);
  const size_t szW_bf  = (size_t)C * C * sizeof(__bf16);

  __bf16* xbf  = (__bf16*)(ws + alloc(szMC_bf));
  __bf16* wqb  = (__bf16*)(ws + alloc(szW_bf));
  __bf16* wkb  = (__bf16*)(ws + alloc(szW_bf));
  __bf16* wvb  = (__bf16*)(ws + alloc(szW_bf));
  __bf16* wob  = (__bf16*)(ws + alloc(szW_bf));
  float*  qpre = (float*)(ws + alloc(szMC_f));   // reused later as ybt
  float*  kpre = (float*)(ws + alloc(szMC_f));
  float*  vpre = (float*)(ws + alloc(szMC_f));
  __bf16* Qhd  = (__bf16*)(ws + alloc(szMC_bf));
  __bf16* Khd  = (__bf16*)(ws + alloc(szMC_bf));
  __bf16* Vtd  = (__bf16*)(ws + alloc(szMC_bf + 256)); // +pad for frag overreads
  __bf16* ybt  = (__bf16*)qpre;                  // alias: qpre dead by then

  // 1) fp32 -> bf16 conversions
  cvt_f32_to_bf16<<<2048, 256, 0, stream>>>(x, xbf, M * C);
  cvt_f32_to_bf16<<<1024, 256, 0, stream>>>(Wq, wqb, (size_t)C * C);
  cvt_f32_to_bf16<<<1024, 256, 0, stream>>>(Wk, wkb, (size_t)C * C);
  cvt_f32_to_bf16<<<1024, 256, 0, stream>>>(Wv, wvb, (size_t)C * C);
  cvt_f32_to_bf16<<<1024, 256, 0, stream>>>(Wo, wob, (size_t)C * C);

  // 2) projection GEMMs: (M x C) = (M x C) @ (C x C)^T
  const int wavesPerGemm = (int)(M / 16) * (C / 64);   // 9312 waves
  const int gemmBlocks = wavesPerGemm / 8;             // 8 waves/block
  gemm_bf16_wmma<<<gemmBlocks, 256, 0, stream>>>(xbf, wqb, qpre, (int)M, C, C);
  gemm_bf16_wmma<<<gemmBlocks, 256, 0, stream>>>(xbf, wkb, kpre, (int)M, C, C);
  gemm_bf16_wmma<<<gemmBlocks, 256, 0, stream>>>(xbf, wvb, vpre, (int)M, C, C);

  // 3) LN + rotary -> (B,H,T,D) bf16
  ln_rope_kernel<<<(int)M, 256, 0, stream>>>(qpre, qg, qb, Qhd, B, T, C);
  ln_rope_kernel<<<(int)M, 256, 0, stream>>>(kpre, kg, kb, Khd, B, T, C);

  // 4) V -> (B,H,D,T) bf16
  v_transpose_kernel<<<2048, 256, 0, stream>>>(vpre, Vtd, B, T, C);

  // 5) flash attention (wave per 16-row q tile)
  const int nQT = (T + 15) / 16;                  // 37
  attn_kernel<<<B * 16 * nQT, 32, 0, stream>>>(Qhd, Khd, Vtd, amask, ybt, B, T);

  // 6) output projection straight into d_out (fp32)
  gemm_bf16_wmma<<<gemmBlocks, 256, 0, stream>>>(ybt, wob, out, (int)M, C, C);
}
